// FlashAttentionV2_9131100471596
// MI455X (gfx1250) — matrix operand order repaired and checked by reading, not compiled, not run on hardware
//
#include <hip/hip_runtime.h>
#include <hip/hip_bf16.h>
#include <math.h>

typedef __attribute__((ext_vector_type(16))) __bf16 v16bf;
typedef __attribute__((ext_vector_type(8)))  float  v8f;

#define NH     16
#define DK     64
#define DMODEL 1024
#define SEQ    2048
#define BATCH  2
#define NROWS  (BATCH * SEQ)   /* 4096 */
#define HALFW  128

union AB16 {
  v16bf v;
  uint4 q[2];
  unsigned short u[16];
};

static __device__ inline unsigned short f2bf(float f) {
  unsigned int u = __float_as_uint(f);
  unsigned int r = (u + 0x7FFFu + ((u >> 16) & 1u)) >> 16;
  return (unsigned short)r;
}

// ---------------- fp32 -> bf16 conversion ----------------
__global__ void convert_f32_bf16(const float* __restrict__ src,
                                 unsigned short* __restrict__ dst, int n) {
  int i = blockIdx.x * blockDim.x + threadIdx.x;
  int stride = gridDim.x * blockDim.x;
  for (; i < n; i += stride) dst[i] = f2bf(src[i]);
}

// ---------------- WMMA GEMM: Y[m,n] = sum_k X[m,k] * W[n,k] ----------------
// Each wave computes a 32x64 output tile: 2 M-subtiles x 4 N-subtiles.
// B operand reused across both M-subtiles; A reused across 4 N-subtiles.
// MODE 0: write bf16 scattered to [B,H,S,DK]       (Q, K; scale folded in)
// MODE 1: write bf16 scattered to [B,H,DK,S]       (V transposed)
// MODE 2: write fp32 row-major   to [NROWS,DMODEL]  (final output)
template<int MODE>
__global__ void __launch_bounds__(256, 1)
gemm32x64(const unsigned short* __restrict__ X,
          const unsigned short* __restrict__ W,
          void* __restrict__ dst, float scale) {
  const int lane = threadIdx.x & 31;
  const int r = lane & 15;
  const int g = lane >> 4;
  const int wave = blockIdx.x * (blockDim.x >> 5) + (threadIdx.x >> 5);
  const int m0 = (wave >> 4) * 32;   // 128 row tiles of 32
  const int n0 = (wave & 15) * 64;   // 16 column quads

  v8f acc[2][4] = {};
  const unsigned short* xrow0 = X + (size_t)(m0 + r) * DMODEL;
  const unsigned short* xrow1 = X + (size_t)(m0 + 16 + r) * DMODEL;

  for (int kk = 0; kk < DMODEL; kk += 32) {
    AB16 a0, a1;
    a0.q[0] = *(const uint4*)(xrow0 + kk + 8 * g);
    a0.q[1] = *(const uint4*)(xrow0 + kk + 16 + 8 * g);
    a1.q[0] = *(const uint4*)(xrow1 + kk + 8 * g);
    a1.q[1] = *(const uint4*)(xrow1 + kk + 16 + 8 * g);
#pragma unroll
    for (int n = 0; n < 4; ++n) {
      AB16 bb;
      const unsigned short* wrow =
          W + (size_t)(n0 + 16 * n + r) * DMODEL + kk + 8 * g;
      bb.q[0] = *(const uint4*)(wrow);
      bb.q[1] = *(const uint4*)(wrow + 16);
      acc[0][n] = __builtin_amdgcn_wmma_f32_16x16x32_bf16(
          false, a0.v, false, bb.v, (short)0, acc[0][n], false, false);
      acc[1][n] = __builtin_amdgcn_wmma_f32_16x16x32_bf16(
          false, a1.v, false, bb.v, (short)0, acc[1][n], false, false);
    }
  }

#pragma unroll
  for (int mi = 0; mi < 2; ++mi) {
#pragma unroll
    for (int n = 0; n < 4; ++n) {
      const int col = n0 + 16 * n + r;
#pragma unroll
      for (int j = 0; j < 8; ++j) {
        const int row = m0 + 16 * mi + j + 8 * g;   // = b*SEQ + s
        const float val = acc[mi][n][j] * scale;
        if (MODE == 2) {
          ((float*)dst)[(size_t)row * DMODEL + col] = val;
        } else {
          const int bb = row / SEQ, s = row % SEQ;
          const int h = col >> 6, d = col & 63;
          size_t addr;
          if (MODE == 0) addr = (((size_t)(bb * NH + h)) * SEQ + s) * DK + d;
          else           addr = (((size_t)(bb * NH + h)) * DK + d) * SEQ + s;
          ((unsigned short*)dst)[addr] = f2bf(val);
        }
      }
    }
  }
}

// ---------------- sliding-window ALiBi flash attention ----------------
// One wave per (b, h, 16-query tile). Scores computed transposed
// (A = K-tile, B = Q^T) so softmax is lane-local + one shfl_xor(16) and
// P lands directly in the A-operand layout for the P*V WMMA.
__global__ void __launch_bounds__(256, 1)
attn_kernel(const unsigned short* __restrict__ Qb,  // [B,NH,S,DK] (pre-scaled)
            const unsigned short* __restrict__ Kb,  // [B,NH,S,DK]
            const unsigned short* __restrict__ Vt,  // [B,NH,DK,S]
            const float* __restrict__ slopes,
            unsigned short* __restrict__ Ob) {      // [B,S,DMODEL] bf16
  const int lane = threadIdx.x & 31;
  const int r = lane & 15;
  const int g = lane >> 4;
  const int wave = blockIdx.x * (blockDim.x >> 5) + (threadIdx.x >> 5);
  const int qt = wave & 127;
  const int h  = (wave >> 7) & (NH - 1);
  const int b  = wave >> 11;
  const int q0 = qt * 16;

  const unsigned short* Qh = Qb + ((size_t)(b * NH + h)) * SEQ * DK;
  const unsigned short* Kh = Kb + ((size_t)(b * NH + h)) * SEQ * DK;
  const unsigned short* Vh = Vt + ((size_t)(b * NH + h)) * DK * SEQ;
  const float slope = slopes[h];

  // Q as B-operand (two K=32 slices), loaded once: b[e] = Q[q0+r, kk+8g+e] (e<8), Q[q0+r, kk+16+8g+e-8] (e>=8)
  AB16 bq0, bq1;
  {
    const unsigned short* qrow = Qh + (size_t)(q0 + r) * DK;
    bq0.q[0] = *(const uint4*)(qrow + 8 * g);
    bq0.q[1] = *(const uint4*)(qrow + 16 + 8 * g);
    bq1.q[0] = *(const uint4*)(qrow + 32 + 8 * g);
    bq1.q[1] = *(const uint4*)(qrow + 48 + 8 * g);
  }

  v8f o[4] = {};
  float m = -1e30f, l = 0.0f;
  const int qpos = q0 + r;

  for (int it = 0; it < 9; ++it) {
    const int key0 = q0 - HALFW + it * 32;   // 32 keys this iteration

    // --- scores: S^T[key, q] ---
    v8f s0 = {}, s1 = {};
    {
      const int row0 = min(max(key0 + r, 0), SEQ - 1);
      const int row1 = min(max(key0 + 16 + r, 0), SEQ - 1);
      const unsigned short* k0p = Kh + (size_t)row0 * DK;
      const unsigned short* k1p = Kh + (size_t)row1 * DK;
      AB16 a0, a1;
      a0.q[0] = *(const uint4*)(k0p + 8 * g);
      a0.q[1] = *(const uint4*)(k0p + 16 + 8 * g);
      a1.q[0] = *(const uint4*)(k0p + 32 + 8 * g);
      a1.q[1] = *(const uint4*)(k0p + 48 + 8 * g);
      s0 = __builtin_amdgcn_wmma_f32_16x16x32_bf16(false, a0.v, false, bq0.v, (short)0, s0, false, false);
      s0 = __builtin_amdgcn_wmma_f32_16x16x32_bf16(false, a1.v, false, bq1.v, (short)0, s0, false, false);
      a0.q[0] = *(const uint4*)(k1p + 8 * g);
      a0.q[1] = *(const uint4*)(k1p + 16 + 8 * g);
      a1.q[0] = *(const uint4*)(k1p + 32 + 8 * g);
      a1.q[1] = *(const uint4*)(k1p + 48 + 8 * g);
      s1 = __builtin_amdgcn_wmma_f32_16x16x32_bf16(false, a0.v, false, bq0.v, (short)0, s1, false, false);
      s1 = __builtin_amdgcn_wmma_f32_16x16x32_bf16(false, a1.v, false, bq1.v, (short)0, s1, false, false);
    }

    // --- ALiBi + window/bounds mask, then tile max (layout [key=j+8g, q=r]) ---
    float tm = -1e30f;
#pragma unroll
    for (int j = 0; j < 8; ++j) {
      const int k0i = key0 + j + 8 * g;
      const int k1i = k0i + 16;
      const int d0 = qpos - k0i, d1 = qpos - k1i;
      const bool v0 = (k0i >= 0) && (k0i < SEQ) && (d0 <= HALFW) && (d0 >= -HALFW);
      const bool v1 = (k1i >= 0) && (k1i < SEQ) && (d1 <= HALFW) && (d1 >= -HALFW);
      s0[j] = v0 ? (s0[j] + slope * (float)d0) : -1e30f;
      s1[j] = v1 ? (s1[j] + slope * (float)d1) : -1e30f;
      tm = fmaxf(tm, fmaxf(s0[j], s1[j]));
    }
    tm = fmaxf(tm, __shfl_xor(tm, 16, 32));
    const float m_new = fmaxf(m, tm);
    const float alpha = __expf(m - m_new);

    // --- exp, pack P to bf16 A-operand, and row-sum, all fused in place ---
    AB16 ap;
    float psum = 0.0f;
#pragma unroll
    for (int j = 0; j < 8; ++j) {
      const float e0 = __expf(s0[j] - m_new);
      const float e1 = __expf(s1[j] - m_new);
      psum += e0 + e1;
      ap.u[j]     = f2bf(e0);
      ap.u[j + 8] = f2bf(e1);
    }
    psum += __shfl_xor(psum, 16, 32);
    l = l * alpha + psum;
    m = m_new;

    // rescale O rows (row q = j+8g lives across VGPRs, alpha lives per lane q=r)
#pragma unroll
    for (int j = 0; j < 8; ++j) {
      const float aj = __shfl(alpha, j + 8 * g, 32);
      o[0][j] *= aj; o[1][j] *= aj; o[2][j] *= aj; o[3][j] *= aj;
    }

    // --- P*V: A operand already lane-local in ap ---
    const bool edge = (key0 < 0) || (key0 + 32 > SEQ);
#pragma unroll
    for (int t = 0; t < 4; ++t) {
      AB16 bv;
      const unsigned short* vrow = Vh + (size_t)(16 * t + r) * SEQ;
      if (!edge) {
        bv.q[0] = *(const uint4*)(vrow + key0 + 8 * g);
        bv.q[1] = *(const uint4*)(vrow + key0 + 16 + 8 * g);
      } else {
#pragma unroll
        for (int e = 0; e < 16; ++e) {
          const int kc = min(max(key0 + 8 * g + e + (e >= 8 ? 8 : 0), 0), SEQ - 1);
          bv.u[e] = vrow[kc];   // P==0 for clamped/invalid keys
        }
      }
      o[t] = __builtin_amdgcn_wmma_f32_16x16x32_bf16(
          false, ap.v, false, bv.v, (short)0, o[t], false, false);
    }
  }

  // --- normalize and store bf16 to [B,S,DMODEL] ---
  const float rl = 1.0f / l;
#pragma unroll
  for (int j = 0; j < 8; ++j) {
    const float linv = __shfl(rl, j + 8 * g, 32);
    const int q = q0 + j + 8 * g;
    const size_t base = ((size_t)(b * SEQ + q)) * DMODEL + h * DK;
#pragma unroll
    for (int t = 0; t < 4; ++t)
      Ob[base + 16 * t + r] = f2bf(o[t][j] * linv);
  }
}

// ---------------- host launch ----------------
extern "C" void kernel_launch(void* const* d_in, const int* in_sizes, int n_in,
                              void* d_out, int out_size, void* d_ws, size_t ws_size,
                              hipStream_t stream) {
  (void)in_sizes; (void)n_in; (void)out_size; (void)ws_size;
  const float* x      = (const float*)d_in[0];
  const float* Wq     = (const float*)d_in[1];
  const float* Wk     = (const float*)d_in[2];
  const float* Wv     = (const float*)d_in[3];
  const float* Wo     = (const float*)d_in[4];
  const float* slopes = (const float*)d_in[5];

  unsigned short* Xb  = (unsigned short*)d_ws;
  unsigned short* Wqb = Xb  + (size_t)NROWS * DMODEL;
  unsigned short* Wkb = Wqb + (size_t)DMODEL * DMODEL;
  unsigned short* Wvb = Wkb + (size_t)DMODEL * DMODEL;
  unsigned short* Wob = Wvb + (size_t)DMODEL * DMODEL;
  unsigned short* Qb  = Wob + (size_t)DMODEL * DMODEL;
  unsigned short* Kb  = Qb  + (size_t)NROWS * DMODEL;
  unsigned short* Vtb = Kb  + (size_t)NROWS * DMODEL;
  unsigned short* Ab  = Vtb + (size_t)NROWS * DMODEL;

  convert_f32_bf16<<<1024, 256, 0, stream>>>(x,  Xb,  NROWS * DMODEL);
  convert_f32_bf16<<<512,  256, 0, stream>>>(Wq, Wqb, DMODEL * DMODEL);
  convert_f32_bf16<<<512,  256, 0, stream>>>(Wk, Wkb, DMODEL * DMODEL);
  convert_f32_bf16<<<512,  256, 0, stream>>>(Wv, Wvb, DMODEL * DMODEL);
  convert_f32_bf16<<<512,  256, 0, stream>>>(Wo, Wob, DMODEL * DMODEL);

  // 2048 waves each (128 M-tiles of 32 x 16 N-quads); 8 waves / 256-thread block
  gemm32x64<0><<<256, 256, 0, stream>>>(Xb, Wqb, (void*)Qb,  0.125f); // Q, 1/sqrt(dk) folded
  gemm32x64<0><<<256, 256, 0, stream>>>(Xb, Wkb, (void*)Kb,  1.0f);   // K
  gemm32x64<1><<<256, 256, 0, stream>>>(Xb, Wvb, (void*)Vtb, 1.0f);   // V transposed

  // 4096 waves: B(2) * NH(16) * 128 query tiles
  attn_kernel<<<512, 256, 0, stream>>>(Qb, Kb, Vtb, slopes, Ab);

  // final projection, fp32 out
  gemm32x64<2><<<256, 256, 0, stream>>>(Ab, Wob, d_out, 1.0f);
}
